// BitFFN_43843026157839
// MI455X (gfx1250) — 8-bit matrix operand order repaired and checked by reading, compiled, not run on hardware
//
#include <hip/hip_runtime.h>
#include <hip/hip_bf16.h>
#include <stdint.h>

typedef __attribute__((ext_vector_type(2))) int v2i;
typedef __attribute__((ext_vector_type(4))) int v4i;
typedef __attribute__((ext_vector_type(8))) int v8i;

#define D_MODEL 2048
#define D_FF    8192
#define NTOK    8192

#if defined(__has_builtin)
#if __has_builtin(__builtin_amdgcn_global_load_async_to_lds_b128)
#define HAVE_ASYNC_LDS 1
#endif
#endif

#ifdef HAVE_ASYNC_LDS
typedef __attribute__((address_space(1))) v4i as1_v4i;
typedef __attribute__((address_space(3))) v4i as3_v4i;

static __device__ __forceinline__ void bitffn_wait_async0() {
#if __has_builtin(__builtin_amdgcn_s_wait_asynccnt)
    __builtin_amdgcn_s_wait_asynccnt(0);
#else
    asm volatile("s_wait_asynccnt 0x0" ::: "memory");
#endif
}
#endif

__global__ void bitffn_absmean_partial(const float* __restrict__ w, size_t n,
                                       float* __restrict__ partial) {
    __shared__ float red[256];
    float s = 0.0f;
    for (size_t i = (size_t)blockIdx.x * 256 + threadIdx.x; i < n;
         i += (size_t)gridDim.x * 256)
        s += fabsf(w[i]);
    red[threadIdx.x] = s;
    __syncthreads();
    for (int st = 128; st > 0; st >>= 1) {
        if ((int)threadIdx.x < st) red[threadIdx.x] += red[threadIdx.x + st];
        __syncthreads();
    }
    if (threadIdx.x == 0) partial[blockIdx.x] = red[0];
}

__global__ void bitffn_absmean_final(const float* __restrict__ partial, int nb,
                                     float inv_n, float* __restrict__ scale_out) {
    __shared__ float red[256];
    float s = 0.0f;
    for (int i = threadIdx.x; i < nb; i += 256) s += partial[i];
    red[threadIdx.x] = s;
    __syncthreads();
    for (int st = 128; st > 0; st >>= 1) {
        if ((int)threadIdx.x < st) red[threadIdx.x] += red[threadIdx.x + st];
        __syncthreads();
    }
    if (threadIdx.x == 0) scale_out[0] = fmaxf(red[0] * inv_n, 1e-8f);
}

__global__ void bitffn_quant_w(const float* __restrict__ w,
                               signed char* __restrict__ wq,
                               const float* __restrict__ scale_ptr, size_t n) {
    const float inv_s = 1.0f / scale_ptr[0];
    for (size_t i = (size_t)blockIdx.x * blockDim.x + threadIdx.x; i < n;
         i += (size_t)gridDim.x * blockDim.x) {
        float q = rintf(w[i] * inv_s);
        q = fminf(1.0f, fmaxf(-1.0f, q));
        wq[i] = (signed char)q;
    }
}

__global__ void bitffn_ln_quant(const float* __restrict__ X, int d,
                                signed char* __restrict__ Xq,
                                float* __restrict__ tokfac) {
    const float* row = X + (size_t)blockIdx.x * d;
    __shared__ float r1[256];
    __shared__ float r2[256];
    __shared__ float sstat[3];
    const int tid = threadIdx.x;

    float s = 0.0f, ss = 0.0f;
    for (int i = tid; i < d; i += 256) {
        float v = row[i];
        s += v;
        ss += v * v;
    }
    r1[tid] = s;
    r2[tid] = ss;
    __syncthreads();
    for (int st = 128; st > 0; st >>= 1) {
        if (tid < st) { r1[tid] += r1[tid + st]; r2[tid] += r2[tid + st]; }
        __syncthreads();
    }
    if (tid == 0) {
        float mu  = r1[0] / (float)d;
        float var = r2[0] / (float)d - mu * mu;
        sstat[0] = mu;
        sstat[1] = rsqrtf(var + 1e-5f);
    }
    __syncthreads();
    const float mu = sstat[0], rstd = sstat[1];

    float amax = 0.0f;
    for (int i = tid; i < d; i += 256)
        amax = fmaxf(amax, fabsf((row[i] - mu) * rstd));
    r1[tid] = amax;
    __syncthreads();
    for (int st = 128; st > 0; st >>= 1) {
        if (tid < st) r1[tid] = fmaxf(r1[tid], r1[tid + st]);
        __syncthreads();
    }
    if (tid == 0) {
        float a = fmaxf(r1[0], 1e-5f);
        sstat[2] = 127.0f / a;
        tokfac[blockIdx.x] = a * (1.0f / 127.0f);
    }
    __syncthreads();
    const float qs = sstat[2];

    signed char* qrow = Xq + (size_t)blockIdx.x * d;
    for (int i = tid; i < d; i += 256) {
        float xn = (row[i] - mu) * rstd;
        float q  = rintf(xn * qs);
        q = fminf(127.0f, fmaxf(-128.0f, q));
        qrow[i] = (signed char)q;
    }
}

template <bool GELU>
__global__ __launch_bounds__(128) void bitffn_gemm_iu8(
    const signed char* __restrict__ Aq,
    const signed char* __restrict__ Wq,
    const float* __restrict__ bias,
    const float* __restrict__ tokfac,
    const float* __restrict__ wscale_ptr,
    float* __restrict__ out,
    int N, int K) {
    const int tid  = threadIdx.x;
    const int lane = tid & 31;
    const int wave = tid >> 5;
    const int l15  = lane & 15;
    const int kh   = lane >> 4;
    const int nBase = blockIdx.x * 128;
    const int mBase = blockIdx.y * 128 + wave * 32;

    __shared__ __align__(16) signed char bs[2][128 * 80];

    const v8i vzero = {0, 0, 0, 0, 0, 0, 0, 0};
    v8i acc[2][8];
#pragma unroll
    for (int mt = 0; mt < 2; ++mt)
#pragma unroll
        for (int nt = 0; nt < 8; ++nt) acc[mt][nt] = vzero;

    const float wscale = wscale_ptr[0];

    auto load_tile = [&](int k0, int buf) {
#pragma unroll
        for (int it = 0; it < 4; ++it) {
            int c   = tid + it * 128;
            int row = c >> 2;
            int off = (c & 3) << 4;
            const signed char* src = Wq + (size_t)(nBase + row) * K + k0 + off;
            signed char*       dst = &bs[buf][row * 80 + off];
#ifdef HAVE_ASYNC_LDS
            __builtin_amdgcn_global_load_async_to_lds_b128(
                (as1_v4i*)(uintptr_t)src,
                (as3_v4i*)(unsigned int)(uintptr_t)dst, 0, 0);
#else
            *(v4i*)dst = *(const v4i*)src;
#endif
        }
    };

    const int nk = K >> 6;

    load_tile(0, 0);
#ifdef HAVE_ASYNC_LDS
    bitffn_wait_async0();
#endif
    __syncthreads();

    for (int i = 0; i < nk; ++i) {
        const int k0  = i << 6;
        const int buf = i & 1;

        if (i + 1 < nk) {
            load_tile(k0 + 64, buf ^ 1);
#ifndef HAVE_ASYNC_LDS
            __builtin_prefetch(Wq + (size_t)(nBase + tid) * K + k0 + 128, 0, 1);
#endif
        }

        v8i a[2];
#pragma unroll
        for (int mt = 0; mt < 2; ++mt) {
            const signed char* ap =
                Aq + (size_t)(mBase + mt * 16 + l15) * K + k0 + kh * 16;
            v8i t = vzero;
#pragma unroll
            for (int g = 0; g < 4; ++g) {
                v2i u = *(const v2i*)(ap + (g >> 1) * 32 + (g & 1) * 8);
                t[2 * g]     = u[0];
                t[2 * g + 1] = u[1];
            }
            a[mt] = t;
        }

#pragma unroll
        for (int nt = 0; nt < 8; ++nt) {
            const signed char* bp = &bs[buf][(nt * 16 + l15) * 80 + kh * 16];
            v4i p0 = *(const v4i*)(bp);
            v4i p1 = *(const v4i*)(bp + 32);
            v8i b;
            b[0] = p0[0]; b[1] = p0[1]; b[2] = p0[2]; b[3] = p0[3];
            b[4] = p1[0]; b[5] = p1[1]; b[6] = p1[2]; b[7] = p1[3];
#pragma unroll
            for (int mt = 0; mt < 2; ++mt)
                acc[mt][nt] = __builtin_amdgcn_wmma_i32_16x16x64_iu8(
                     true, a[mt],  true, b, acc[mt][nt],
                     false,  false);
        }

#ifdef HAVE_ASYNC_LDS
        bitffn_wait_async0();
#endif
        __syncthreads();
    }

#pragma unroll
    for (int mt = 0; mt < 2; ++mt) {
#pragma unroll
        for (int j = 0; j < 8; ++j) {
            const int token = mBase + mt * 16 + kh * 8 + j;
            const float f   = tokfac[token] * wscale;
#pragma unroll
            for (int nt = 0; nt < 8; ++nt) {
                const int feat = nBase + nt * 16 + l15;
                float v = (float)acc[mt][nt][j] * f + bias[feat];
                if (GELU)
                    v = 0.5f * v * (1.0f + erff(v * 0.70710678118654752440f));
                out[(size_t)token * N + feat] = v;
            }
        }
    }
}

extern "C" void kernel_launch(void* const* d_in, const int* in_sizes, int n_in,
                              void* d_out, int out_size, void* d_ws, size_t ws_size,
                              hipStream_t stream) {
    (void)in_sizes; (void)n_in; (void)out_size; (void)ws_size;
    const float* x  = (const float*)d_in[0];
    const float* w1 = (const float*)d_in[1];
    const float* b1 = (const float*)d_in[2];
    const float* w2 = (const float*)d_in[3];
    const float* b2 = (const float*)d_in[4];

    const size_t NW1 = (size_t)D_FF * D_MODEL;
    const size_t NW2 = (size_t)D_MODEL * D_FF;
    const size_t NX  = (size_t)NTOK * D_MODEL;
    const size_t NH  = (size_t)NTOK * D_FF;

    char* p = (char*)d_ws;
    signed char* w1q = (signed char*)p;  p += NW1;
    signed char* w2q = (signed char*)p;  p += NW2;
    signed char* xq  = (signed char*)p;  p += NX;
    signed char* hq  = (signed char*)p;  p += NH;
    float* h    = (float*)p;             p += NH * sizeof(float);
    float* xfac = (float*)p;             p += NTOK * sizeof(float);
    float* hfac = (float*)p;             p += NTOK * sizeof(float);
    float* part1 = (float*)p;            p += 1024 * sizeof(float);
    float* part2 = (float*)p;            p += 1024 * sizeof(float);
    float* w1s  = (float*)p;             p += 16;
    float* w2s  = (float*)p;             p += 16;

    bitffn_absmean_partial<<<1024, 256, 0, stream>>>(w1, NW1, part1);
    bitffn_absmean_partial<<<1024, 256, 0, stream>>>(w2, NW2, part2);
    bitffn_absmean_final<<<1, 256, 0, stream>>>(part1, 1024, 1.0f / (float)NW1, w1s);
    bitffn_absmean_final<<<1, 256, 0, stream>>>(part2, 1024, 1.0f / (float)NW2, w2s);

    bitffn_quant_w<<<4096, 256, 0, stream>>>(w1, w1q, w1s, NW1);
    bitffn_quant_w<<<4096, 256, 0, stream>>>(w2, w2q, w2s, NW2);

    bitffn_ln_quant<<<NTOK, 256, 0, stream>>>(x, D_MODEL, xq, xfac);

    bitffn_gemm_iu8<true><<<dim3(D_FF / 128, NTOK / 128), 128, 0, stream>>>(
        xq, w1q, b1, xfac, w1s, h, D_FF, D_MODEL);

    bitffn_ln_quant<<<NTOK, 256, 0, stream>>>(h, D_FF, hq, hfac);

    bitffn_gemm_iu8<false><<<dim3(D_MODEL / 128, NTOK / 128), 128, 0, stream>>>(
        hq, w2q, b2, hfac, w2s, (float*)d_out, D_MODEL, D_FF);
}
